// ROIEntropySmoothL1Loss_45234595561955
// MI455X (gfx1250) — matrix operand order repaired and checked by reading, not compiled, not run on hardware
//
#include <hip/hip_runtime.h>

// ---------------------------------------------------------------------------
// ROI entropy-refined stereo teacher loss, fused for MI455X (gfx1250).
//  - one wave32-workgroup per (sample,row); only ROI rows launched (54/160)
//  - banded cost volume (d=0..64) via v_wmma_f32_16x16x32_f16, f32 accum
//  - 65x576 f32 cost band (+2 guard rows) + f16 R-row in 320KB WGP LDS
//  - guard-row clamp makes all WMMA D-scatter stores unconditional
//  - readfirstlane makes tile loops scalar: EXEC all-ones around WMMA
//  - entropy, run extents, prefix-sum refinement, argmax, smooth-L1 fused
//  - deterministic: per-row partials in ws, tree reductions, no float atomics
// ---------------------------------------------------------------------------

typedef _Float16 v16h __attribute__((ext_vector_type(16)));
typedef _Float16 v4h  __attribute__((ext_vector_type(4)));
typedef float    v8f  __attribute__((ext_vector_type(8)));
typedef float    v4f  __attribute__((ext_vector_type(4)));

#define B_N   8
#define H_N   160
#define W_N   576
#define C_N   96
#define D_N   65          // MAX_DISP + 1
#define ROW0  106         // floor(2/3 * 160)
#define NROWS 54          // rows 106..159
#define U0    172         // floor(0.3 * 576)
#define U1    403         // ceil(0.7 * 576) - 1
#define NBLK  (B_N * NROWS)
#define EPS_F 1e-8f
// -EPS*log(EPS) contribution of each masked (-inf) softmax entry
#define EPS_ENT 1.8420680743952367e-7f

// dynamic-LDS layout (bytes)
#define LDS_GLO_OFF  0                               // guard row (clamped d = -1)
#define LDS_COST_OFF (W_N * 4)                       // 2304: cost rows d = 0..64
#define LDS_GHI_OFF  (LDS_COST_OFF + D_N * W_N * 4)  // 152064: guard row (d = 65)
#define LDS_RF_OFF   (LDS_GHI_OFF + W_N * 4)         // 154368: f16 R row [576][96]
#define LDS_VAL_OFF  (LDS_RF_OFF + W_N * C_N * 2)    // 264960
#define LDS_A_OFF    (LDS_VAL_OFF + W_N)             // 265536
#define LDS_B_OFF    (LDS_A_OFF + W_N * 2)           // 266688
#define LDS_TOTAL    (LDS_B_OFF + W_N * 2)           // 267840 < 320KB WGP LDS

__device__ __forceinline__ int imin(int a, int b) { return a < b ? a : b; }
__device__ __forceinline__ int imax(int a, int b) { return a > b ? a : b; }

__global__ void __launch_bounds__(256)
roi_entropy_row_kernel(const float* __restrict__ FqL,
                       const float* __restrict__ FqR,
                       const float* __restrict__ sdisp,
                       float* __restrict__ rowAcc)
{
    extern __shared__ char smem[];
    float*         costG  = (float*)(smem + LDS_GLO_OFF);    // [67][576] incl guards
    float*         cost   = (float*)(smem + LDS_COST_OFF);   // [65][576]
    _Float16*      rf     = (_Float16*)(smem + LDS_RF_OFF);  // [576][96] f16 R row
    unsigned char* validA = (unsigned char*)(smem + LDS_VAL_OFF);
    short*         aArr   = (short*)(smem + LDS_A_OFF);
    short*         bArr   = (short*)(smem + LDS_B_OFF);
    __shared__ float redL[256];
    __shared__ float redC[256];

    const int blk = blockIdx.x;            // 0..431
    const int bi  = blk / NROWS;
    const int h   = ROW0 + (blk % NROWS);
    const int tid = threadIdx.x;

    const size_t rowOff = ((size_t)bi * H_N + h) * (size_t)W_N * C_N;
    const float* Lrow = FqL + rowOff;
    const float* Rrow = FqR + rowOff;

    // ---- stage R row into LDS as f16 (f32x4 reads, packed f16x4 stores) ----
    for (int idx = tid; idx < (W_N * C_N) / 4; idx += 256) {
        v4f v = ((const v4f*)Rrow)[idx];           // global_load_b128
        v4h hv;
        hv[0] = (_Float16)v.x; hv[1] = (_Float16)v.y;
        hv[2] = (_Float16)v.z; hv[3] = (_Float16)v.w;
        *(v4h*)(rf + idx * 4) = hv;                // ds_store_b64
    }
    __syncthreads();

    // ---- banded cost volume via WMMA --------------------------------------
    // Tile (i,j): L pixels w in [16i,16i+15], R pixels w' in [16j,16j+15],
    // keep d = w - w' in [0,64]  =>  j in [i-4, i].
    const int wave = __builtin_amdgcn_readfirstlane(tid >> 5);  // scalar loops
    const int lane = tid & 31;
    const int l15  = lane & 15;
    const int lhi  = lane >> 4;                    // 0 or 1
    const int koff = lhi ? 16 : 0;                 // A-frag K split: contiguous k = 16*lhi + s

    for (int i = wave; i < 36; i += 8) {           // scalar, EXEC all-ones
        const int wA = 16 * i + l15;
        const float* lp = Lrow + (size_t)wA * C_N;
        v16h Afr[3];
#pragma unroll
        for (int k = 0; k < 3; ++k) {
            // lane<16: K = 32k+0..15; lane>=16: K = 32k+16..31
            const float* p = lp + 32 * k + koff;
            v4f f0 = *(const v4f*)(p);
            v4f f1 = *(const v4f*)(p + 4);
            v4f f2 = *(const v4f*)(p + 8);
            v4f f3 = *(const v4f*)(p + 12);
            v16h a;
            a[0]  = (_Float16)f0.x; a[1]  = (_Float16)f0.y;
            a[2]  = (_Float16)f0.z; a[3]  = (_Float16)f0.w;
            a[4]  = (_Float16)f1.x; a[5]  = (_Float16)f1.y;
            a[6]  = (_Float16)f1.z; a[7]  = (_Float16)f1.w;
            a[8]  = (_Float16)f2.x; a[9]  = (_Float16)f2.y;
            a[10] = (_Float16)f2.z; a[11] = (_Float16)f2.w;
            a[12] = (_Float16)f3.x; a[13] = (_Float16)f3.y;
            a[14] = (_Float16)f3.z; a[15] = (_Float16)f3.w;
            Afr[k] = a;
        }
        const int jmin = imax(0, i - 4);
        for (int j = jmin; j <= i; ++j) {          // scalar loop
            const int w2 = 16 * j + l15;           // B column = R pixel
            // B-frag: lane<16 -> K 32k+0..15, lane>=16 -> K 32k+16..31
            const _Float16* bp = rf + (size_t)w2 * C_N + 16 * lhi;
            v16h b0 = *(const v16h*)(bp);          // 3x 32B LDS reads, batched
            v16h b1 = *(const v16h*)(bp + 32);
            v16h b2 = *(const v16h*)(bp + 64);
            v8f acc = {0.f, 0.f, 0.f, 0.f, 0.f, 0.f, 0.f, 0.f};
            acc = __builtin_amdgcn_wmma_f32_16x16x32_f16(
                false, Afr[0], false, b0, (short)0, acc, false, false);
            acc = __builtin_amdgcn_wmma_f32_16x16x32_f16(
                false, Afr[1], false, b1, (short)0, acc, false, false);
            acc = __builtin_amdgcn_wmma_f32_16x16x32_f16(
                false, Afr[2], false, b2, (short)0, acc, false, false);
            // D scatter: C/D layout M = r + 8*lhi, N = l15; d = wm - w2.
            // Clamp d to [-1,65]: out-of-band lanes write unread guard rows,
            // keeping every store unconditional (no exec masking, no hazards).
            const int dbase = 16 * (i - j) + 8 * lhi - l15;   // d at r = 0
            const int wm0   = 16 * i + 8 * lhi;
#pragma unroll
            for (int r = 0; r < 8; ++r) {
                const int dc = imin(imax(dbase + r, -1), D_N);
                costG[(dc + 1) * W_N + (wm0 + r)] = acc[r];
            }
        }
    }
    __syncthreads();

    // ---- entropy map per column (softmax over d, T = 0.1) ------------------
    for (int w = tid; w < W_N; w += 256) {
        const int dmax = imin(w, D_N - 1);
        float m = cost[w];
        for (int d = 1; d <= dmax; ++d) m = fmaxf(m, cost[d * W_N + w]);
        float s = 0.f;
        for (int d = 0; d <= dmax; ++d)
            s += __expf((cost[d * W_N + w] - m) * 10.0f);
        float ent = 0.f;
        for (int d = 0; d <= dmax; ++d) {
            float e = __expf((cost[d * W_N + w] - m) * 10.0f);
            float p = fmaxf(e / s, EPS_F);
            ent -= p * __logf(p);
        }
        ent += (float)(D_N - 1 - dmax) * EPS_ENT;   // masked (-inf) entries: p=EPS
        const int Deff = dmax + 1;
        ent = (Deff > 1) ? ent / (__logf((float)Deff) + EPS_F) : 0.f;
        ent = fminf(fmaxf(ent, 0.f), 1.f);
        validA[w] = (ent <= 0.65f) ? 1 : 0;
    }
    __syncthreads();

    // ---- run extents (serial per row; 576 steps, negligible) ---------------
    if (tid == 0) {
        int prev = -1;
        for (int w = 0; w < W_N; ++w) {
            if (validA[w]) prev = w;
            aArr[w] = (short)(prev + 1);            // a = prev_valid + 1 (0 if none)
        }
        int nxt = W_N;
        for (int w = W_N - 1; w >= 0; --w) {
            if (validA[w]) nxt = w;
            bArr[w] = (short)imin(nxt - 1, W_N - 1); // b = next_valid - 1 (W-1 if none)
        }
    }
    __syncthreads();

    // ---- in-place inclusive prefix sums along w for each disparity ---------
    if (tid < D_N) {
        float run = 0.f;
        float* cr = cost + tid * W_N;
        for (int w = 0; w < W_N; ++w) {
            run += (w >= tid) ? cr[w] : 0.f;        // finite <=> w >= d
            cr[w] = run;
        }
    }
    __syncthreads();

    // ---- refinement + argmax teacher + smooth-L1 on accepted pixels --------
    float lsum = 0.f, lcnt = 0.f;
    for (int w = U0 + tid; w <= U1; w += 256) {
        if (!validA[w]) {                           // accept = invalid & roi
            const int a = (int)aArr[w];
            const int b = (int)bArr[w];
            float best = -1e30f;
            int bestd = 0;
            for (int d = 0; d < D_N; ++d) {
                const int lo = imax(a, d);
                if (b >= lo) {                      // den > 0
                    const float* cr = cost + d * W_N;
                    float num = cr[b] - ((lo > 0) ? cr[lo - 1] : 0.f);
                    float agg = num / (float)(b - lo + 1);
                    if (agg > best) { best = agg; bestd = d; }  // first-max tiebreak
                }
            }
            float s  = sdisp[((size_t)bi * H_N + h) * W_N + w];
            float ad = fabsf(s - (float)bestd);
            float sl1 = (ad < 1.f) ? 0.5f * ad * ad : ad - 0.5f;  // BETA = 1
            lsum += sl1;
            lcnt += 1.f;
        }
    }
    redL[tid] = lsum;
    redC[tid] = lcnt;
    __syncthreads();
    for (int s = 128; s > 0; s >>= 1) {             // deterministic tree reduce
        if (tid < s) { redL[tid] += redL[tid + s]; redC[tid] += redC[tid + s]; }
        __syncthreads();
    }
    if (tid == 0) {
        rowAcc[blk * 2 + 0] = redL[0];
        rowAcc[blk * 2 + 1] = redC[0];
    }
}

__global__ void roi_entropy_reduce_kernel(const float* __restrict__ rowAcc,
                                          float* __restrict__ out)
{
    __shared__ float ps[B_N];
    const int t = threadIdx.x;
    if (t < B_N) {
        float s = 0.f, c = 0.f;
        for (int r = 0; r < NROWS; ++r) {
            s += rowAcc[(t * NROWS + r) * 2 + 0];
            c += rowAcc[(t * NROWS + r) * 2 + 1];
        }
        ps[t] = s / fmaxf(c, 1.f);
    }
    __syncthreads();
    if (t == 0) {
        float m = 0.f;
        for (int i = 0; i < B_N; ++i) m += ps[i];
        out[0] = m / (float)B_N;
    }
}

extern "C" void kernel_launch(void* const* d_in, const int* in_sizes, int n_in,
                              void* d_out, int out_size, void* d_ws, size_t ws_size,
                              hipStream_t stream)
{
    (void)in_sizes; (void)n_in; (void)out_size; (void)ws_size;
    const float* FqL = (const float*)d_in[0];
    const float* FqR = (const float*)d_in[1];
    const float* sd  = (const float*)d_in[2];
    float* out    = (float*)d_out;
    float* rowAcc = (float*)d_ws;                  // 432 * 2 floats

    // 267840 B dynamic LDS per workgroup (fits 320KB WGP LDS); allow > default
    hipFuncSetAttribute((const void*)roi_entropy_row_kernel,
                        hipFuncAttributeMaxDynamicSharedMemorySize, LDS_TOTAL);

    roi_entropy_row_kernel<<<dim3(NBLK), dim3(256), LDS_TOTAL, stream>>>(
        FqL, FqR, sd, rowAcc);
    roi_entropy_reduce_kernel<<<dim3(1), dim3(32), 0, stream>>>(rowAcc, out);
}
